// KMeans_17978733101581
// MI455X (gfx1250) — matrix operand order repaired and checked by reading, not compiled, not run on hardware
//
#include <hip/hip_runtime.h>
#include <hip/hip_bf16.h>

// MI455X (gfx1250) K-means assignment.
// dist^2[m,n] = x2[m] - 2*X[m]·C[n] + c2[n]; argmin_n, then sqrt(min).
// GEMM via v_wmma_f32_16x16x32_bf16 with error-compensated hi/lo bf16 split
// of both operands (Hi*Hi + Hi*Lo + Lo*Hi ~ fp32 accuracy), X pre-scaled by
// -2 so the accumulator directly holds -2*dot. Codebook staged in LDS as
// interleaved 304-byte row blocks [hi(128B)|pad|lo(128B)|pad|c2(4B)|pad]:
//  - single base VGPR + immediate offsets for all 9 DS accesses per tile
//  - 304B = 76 dwords = 12 mod 64 banks -> conflict-free ds_load_b128
// Accumulator chain starts from inline 0; c2 added in the compare epilogue.

typedef __attribute__((ext_vector_type(16))) __bf16 v16bf;
typedef __attribute__((ext_vector_type(8)))  float  v8f;

#define KN 512
#define WDIM 64
#define ROW_BYTES 304         // hi[0,128) pad lo[144,272) pad c2[288,292) pad
#define LO_OFF 144
#define C2_OFF 288
#define ROWS_PER_WG 128
#define THREADS 256

__device__ __forceinline__ __bf16 f2bf(float f) {
  unsigned u = __float_as_uint(f);
  u += 0x7FFFu + ((u >> 16) & 1u);            // round-to-nearest-even
  unsigned short s = (unsigned short)(u >> 16);
  return __builtin_bit_cast(__bf16, s);
}
__device__ __forceinline__ float bf2f(__bf16 b) {
  unsigned short s = __builtin_bit_cast(unsigned short, b);
  return __uint_as_float(((unsigned)s) << 16);
}

// Convert 8 floats (scaled by -2) into elements [p, p+8) of hi/lo A fragments.
__device__ __forceinline__ void cvt8(const float4& a, const float4& b,
                                     v16bf& hi, v16bf& lo, int p) {
  float v[8] = {a.x, a.y, a.z, a.w, b.x, b.y, b.z, b.w};
#pragma unroll
  for (int j = 0; j < 8; ++j) {
    float t = -2.0f * v[j];
    __bf16 h = f2bf(t);
    float r = t - bf2f(h);
    hi[p + j] = h;
    lo[p + j] = f2bf(r);
  }
}

__global__ __launch_bounds__(THREADS) void kmeans_assign_kernel(
    const float* __restrict__ X, const float* __restrict__ CB,
    int* __restrict__ outIdx, float* __restrict__ outDist, int n_rows) {
  extern __shared__ char smem[];  // KN * ROW_BYTES interleaved row blocks

  const int tid = threadIdx.x;

  // ---- Stage codebook into LDS: per-row [bf16 hi | bf16 lo | f32 c2] ----
  for (int e = tid * 4; e < KN * WDIM; e += THREADS * 4) {
    float4 v = *(const float4*)(CB + e);
    int n = e >> 6, w = e & 63;
    char* rb = smem + n * ROW_BYTES;
    __bf16* rowHi = (__bf16*)rb + w;
    __bf16* rowLo = (__bf16*)(rb + LO_OFF) + w;
    float vv[4] = {v.x, v.y, v.z, v.w};
#pragma unroll
    for (int j = 0; j < 4; ++j) {
      __bf16 h = f2bf(vv[j]);
      rowHi[j] = h;
      rowLo[j] = f2bf(vv[j] - bf2f(h));
    }
  }
  for (int n = tid; n < KN; n += THREADS) {
    const float4* r = (const float4*)(CB + n * WDIM);
    float s = 0.f;
#pragma unroll
    for (int i = 0; i < WDIM / 4; ++i) {
      float4 v = r[i];
      s += v.x * v.x + v.y * v.y + v.z * v.z + v.w * v.w;
    }
    *(float*)(smem + n * ROW_BYTES + C2_OFF) = s;  // full-precision c2[n]
  }
  __syncthreads();

  // ---- Per-wave 16-row slab ----
  const int wave = tid >> 5;
  const int lane = tid & 31;
  const int l16  = lane & 15;
  const bool hiHalf = lane >= 16;
  const long base = (long)blockIdx.x * ROWS_PER_WG + (long)wave * 16;
  long row = base + l16;
  if (row >= n_rows) row = n_rows - 1;  // clamp; OOB rows computed, not stored
  const float* xr = X + row * WDIM;
  const int g0 = hiHalf ? 16 : 0;

  // A-matrix 16x32 bf16 layout: lane L holds row M=L%16;
  // lanes<16: w 0..15 (+32 chunk1); lanes>=16: w 16..31.
  float4 G[8];
#pragma unroll
  for (int g = 0; g < 4; ++g) {
    G[2 * g]     = *(const float4*)(xr + g0 + 32 * (g >> 1) + 8 * (g & 1));
    G[2 * g + 1] = *(const float4*)(xr + g0 + 32 * (g >> 1) + 8 * (g & 1) + 4);
  }
  float part = 0.f;
#pragma unroll
  for (int i = 0; i < 8; ++i)
    part += G[i].x * G[i].x + G[i].y * G[i].y + G[i].z * G[i].z + G[i].w * G[i].w;
  const float x2full = part + __shfl_xor(part, 16, 32);  // full ||x||^2 of row l16

  v16bf aHi0{}, aLo0{}, aHi1{}, aLo1{};
  cvt8(G[0], G[1], aHi0, aLo0, 0);   // chunk0: w g0..g0+7
  cvt8(G[2], G[3], aHi0, aLo0, 8);   // chunk0: w g0+8..g0+15
  cvt8(G[4], G[5], aHi1, aLo1, 0);   // chunk1: +32
  cvt8(G[6], G[7], aHi1, aLo1, 8);   // chunk1: w g0+40..g0+47

  v8f bestS;
  int bestN[8];
#pragma unroll
  for (int i = 0; i < 8; ++i) { bestS[i] = 3.0e38f; bestN[i] = 0; }

  // B-matrix 32x16 bf16 layout: lane = column n (N=lane%16),
  // lanes<16 hold K(w)=0..15, lanes>=16 hold w=16..31 (contiguous in w).
  const int woff2 = hiHalf ? 32 : 0;                 // byte shift into row block
  const char* p  = smem + l16 * ROW_BYTES + woff2;   // B fragments (induction)
  const char* pc = smem + l16 * ROW_BYTES + C2_OFF;  // c2 (induction)
  const v8f zeroC = {};                              // lowers to inline SRC2=0

#pragma unroll 4
  for (int t = 0; t < KN / 16; ++t) {
    v16bf bHi0 = *(const v16bf*)(p);             // hi, chunk0
    v16bf bHi1 = *(const v16bf*)(p + 64);        // hi, chunk1
    v16bf bLo0 = *(const v16bf*)(p + LO_OFF);    // lo, chunk0
    v16bf bLo1 = *(const v16bf*)(p + LO_OFF + 64);
    const float c2v = *(const float*)(pc);
    p  += 16 * ROW_BYTES;
    pc += 16 * ROW_BYTES;

    // acc = -2 * x·c (A pre-scaled by -2), split-precision, single chain
    v8f acc;
    acc = __builtin_amdgcn_wmma_f32_16x16x32_bf16(false, aHi0, false, bHi0, (short)0, zeroC, false, false);
    acc = __builtin_amdgcn_wmma_f32_16x16x32_bf16(false, aHi1, false, bHi1, (short)0, acc, false, false);
    acc = __builtin_amdgcn_wmma_f32_16x16x32_bf16(false, aLo0, false, bHi0, (short)0, acc, false, false);
    acc = __builtin_amdgcn_wmma_f32_16x16x32_bf16(false, aHi0, false, bLo0, (short)0, acc, false, false);
    acc = __builtin_amdgcn_wmma_f32_16x16x32_bf16(false, aLo1, false, bHi1, (short)0, acc, false, false);
    acc = __builtin_amdgcn_wmma_f32_16x16x32_bf16(false, aHi1, false, bLo1, (short)0, acc, false, false);

    const int n = t * 16 + l16;
#pragma unroll
    for (int i = 0; i < 8; ++i) {
      float s = acc[i] + c2v;                        // s = c2 - 2*dot
      if (s < bestS[i]) { bestS[i] = s; bestN[i] = n; }  // strict <: first-min wins
    }
  }

  // ---- Cross-lane argmin within each 16-lane half (D: N = lane%16) ----
#pragma unroll
  for (int off = 8; off >= 1; off >>= 1) {
#pragma unroll
    for (int i = 0; i < 8; ++i) {
      float oS = __shfl_xor(bestS[i], off, 32);
      int   oN = __shfl_xor(bestN[i], off, 32);
      if (oS < bestS[i] || (oS == bestS[i] && oN < bestN[i])) {
        bestS[i] = oS;
        bestN[i] = oN;
      }
    }
  }

  // D layout: VGPR r = row r (lanes<16) / row 8+r (lanes>=16).
  // Writer lanes: row r -> lane r (has x2 of row r); row 8+r -> lane 24+r.
#pragma unroll
  for (int r = 0; r < 8; ++r) {
    long wrow = base + ((lane < 16) ? r : (8 + r));
    bool active = (lane == r) || (lane == 24 + r);
    if (active && wrow < n_rows) {
      float d2 = fmaxf(x2full + bestS[r], 0.0f);
      outIdx[wrow]  = bestN[r];
      outDist[wrow] = sqrtf(d2);
    }
  }
}

extern "C" void kernel_launch(void* const* d_in, const int* in_sizes, int n_in,
                              void* d_out, int out_size, void* d_ws, size_t ws_size,
                              hipStream_t stream) {
  const float* X  = (const float*)d_in[0];   // [N, 64]
  const float* CB = (const float*)d_in[1];   // [512, 64]
  const int n_rows = in_sizes[0] / WDIM;     // 500000
  int*   outIdx  = (int*)d_out;              // idx first (reference return order)
  float* outDist = (float*)d_out + n_rows;   // then min_dists

  const int grid = (n_rows + ROWS_PER_WG - 1) / ROWS_PER_WG;
  const size_t smem = (size_t)KN * ROW_BYTES;  // 152 KB -> 2 workgroups per WGP
  kmeans_assign_kernel<<<grid, THREADS, smem, stream>>>(X, CB, outIdx, outDist, n_rows);
}
